// Seq2SeqModel_24446953849219
// MI455X (gfx1250) — matrix operand order repaired and checked by reading, not compiled, not run on hardware
//
#include <hip/hip_runtime.h>
#include <hip/hip_bf16.h>

typedef __bf16 bf16;
typedef __attribute__((ext_vector_type(16))) __bf16 v16bf;
typedef __attribute__((ext_vector_type(8)))  __bf16 v8bf;
typedef __attribute__((ext_vector_type(8)))  float  v8f;

// ---------------------------------------------------------------------------
// WMMA fragment helpers (CDNA5 wave32 layouts, CDNA5 ISA §7.12.2)
// ---------------------------------------------------------------------------

// A-matrix 16x32 bf16: lane m = lane&15, khalf = lane>>4.
// elems 0..7  -> K = k0 + khalf*8 + {0..7}
// elems 8..15 -> K = k0 + 16 + khalf*8 + {0..7}
__device__ inline v16bf load_a_frag(const bf16* __restrict__ A, int lda,
                                    int m0, int k0, int lane) {
  int m  = m0 + (lane & 15);
  int kh = (lane >> 4) * 8;
  const bf16* p = A + (size_t)m * lda + k0 + kh;
  union { v16bf v; v8bf h[2]; } u;
  u.h[0] = *(const v8bf*)(p);
  u.h[1] = *(const v8bf*)(p + 16);
  return u.v;
}

// B-matrix 32x16 bf16 from pre-transposed weights Bt[N][K]:
// lane col n = lane&15, khalf = lane>>4; elems 0..7 -> K = k0 + khalf*8 + {0..7},
// elems 8..15 -> K = k0 + 16 + khalf*8 + {0..7}
__device__ inline v16bf load_b_frag(const bf16* __restrict__ Bt, int ldb,
                                    int n0, int k0, int lane) {
  int n  = n0 + (lane & 15);
  int kh = (lane >> 4) * 8;
  const bf16* p = Bt + (size_t)n * ldb + k0 + kh;
  union { v16bf v; v8bf h[2]; } u;
  u.h[0] = *(const v8bf*)(p);
  u.h[1] = *(const v8bf*)(p + 16);
  return u.v;
}

__device__ inline v8f wmma_bf16(v16bf a, v16bf b, v8f c) {
  return __builtin_amdgcn_wmma_f32_16x16x32_bf16(
      /*neg_a=*/false, a, /*neg_b=*/false, b,
      /*c_mod=*/(short)0, c, /*reuse_a=*/false, /*reuse_b=*/false);
}

__device__ inline float sigmoidf_(float x) { return 1.0f / (1.0f + __expf(-x)); }

__device__ inline void store_tile(v8f acc, float bv, int act, int mrow, int n,
                                  float* __restrict__ Cf, long ldcf, long cbase,
                                  bf16* __restrict__ Cb, int ldcb) {
#pragma unroll
  for (int r = 0; r < 8; ++r) {
    float v = acc[r] + bv;
    if (act == 1) v = fmaxf(v, 0.0f);
    int m = mrow + r;
    if (Cf) Cf[cbase + (long)m * ldcf + n] = v;
    if (Cb) Cb[(size_t)m * ldcb + n] = (bf16)v;
  }
}

// ---------------------------------------------------------------------------
// Generic bf16 WMMA GEMM:  C[M][N] = act(A[M][K] @ Bt[N][K]^T + bias)
// Register-blocked: each wave owns one 16-col n-tile and ALL four m-tiles
// (4 accumulators), so each K-step is 1 B-frag + 4 A-frags -> 4 WMMAs.
// block = 8 waves -> 64(M) x 128(N) tile; grid = (N/128, M/64).
// A-frags are identical across the block's waves (WGP-cache hits); B is
// loaded exactly once per block.
// ---------------------------------------------------------------------------
__global__ void __launch_bounds__(256)
gemm_kernel(const bf16* __restrict__ A, int lda,
            const bf16* __restrict__ Bt, int ldb,
            const float* __restrict__ bias,
            float* __restrict__ Cf, long ldcf, long cbase,
            bf16* __restrict__ Cb, int ldcb,
            int K, int act) {
  int lane = threadIdx.x;
  int w    = threadIdx.y;
  int m0 = blockIdx.y * 64;
  int n0 = blockIdx.x * 128 + w * 16;
  v8f acc0 = {}, acc1 = {}, acc2 = {}, acc3 = {};
  for (int k0 = 0; k0 < K; k0 += 32) {
    v16bf b  = load_b_frag(Bt, ldb, n0, k0, lane);
    v16bf a0 = load_a_frag(A, lda, m0 +  0, k0, lane);
    v16bf a1 = load_a_frag(A, lda, m0 + 16, k0, lane);
    v16bf a2 = load_a_frag(A, lda, m0 + 32, k0, lane);
    v16bf a3 = load_a_frag(A, lda, m0 + 48, k0, lane);
    acc0 = wmma_bf16(a0, b, acc0);
    acc1 = wmma_bf16(a1, b, acc1);
    acc2 = wmma_bf16(a2, b, acc2);
    acc3 = wmma_bf16(a3, b, acc3);
  }
  int n  = n0 + (lane & 15);
  int mh = (lane >> 4) << 3;
  float bv = bias ? bias[n] : 0.0f;
  store_tile(acc0, bv, act, m0 +  0 + mh, n, Cf, ldcf, cbase, Cb, ldcb);
  store_tile(acc1, bv, act, m0 + 16 + mh, n, Cf, ldcf, cbase, Cb, ldcb);
  store_tile(acc2, bv, act, m0 + 32 + mh, n, Cf, ldcf, cbase, Cb, ldcb);
  store_tile(acc3, bv, act, m0 + 48 + mh, n, Cf, ldcf, cbase, Cb, ldcb);
}

// ---------------------------------------------------------------------------
// Fused GRU step: gh = h @ Wr (3 gate tiles share one A-fragment), then gates.
// gx (x@Wk + b0) is precomputed. grid = (U/16, ndir), block = (32,4).
// Each wave: one M-tile (16 batch rows) x 16 u-cols, 3 accumulators (z,r,n).
// ---------------------------------------------------------------------------
struct GruDir {
  const bf16*  h_in_bf;   // [64][1024]
  const float* h_in_f;    // [64][1024]
  const bf16*  Wrt;       // [3072][1024] transposed bf16
  const float* b1;        // bias row 1, [3072]
  const float* gx;        // [64][3072] (x@Wk + b0) for this timestep
  float*       h_out_f;
  bf16*        h_out_bf;
  bf16*        y;         // [64][1024] for this timestep, or null
};

__global__ void __launch_bounds__(128)
gru_step_kernel(GruDir p0, GruDir p1) {
  GruDir p = (blockIdx.y == 0) ? p0 : p1;
  int lane = threadIdx.x;
  int m0 = threadIdx.y * 16;
  int u0 = blockIdx.x * 16;
  v8f az = {}, ar = {}, an = {};
  for (int k0 = 0; k0 < 1024; k0 += 32) {
    v16bf a  = load_a_frag(p.h_in_bf, 1024, m0, k0, lane);
    v16bf bz = load_b_frag(p.Wrt, 1024, u0,        k0, lane);
    v16bf br = load_b_frag(p.Wrt, 1024, u0 + 1024, k0, lane);
    v16bf bn = load_b_frag(p.Wrt, 1024, u0 + 2048, k0, lane);
    az = wmma_bf16(a, bz, az);
    ar = wmma_bf16(a, br, ar);
    an = wmma_bf16(a, bn, an);
  }
  int u  = u0 + (lane & 15);
  int mb = m0 + ((lane >> 4) << 3);
  float b1z = p.b1[u], b1r = p.b1[1024 + u], b1n = p.b1[2048 + u];
#pragma unroll
  for (int r = 0; r < 8; ++r) {
    int b = mb + r;
    const float* gxrow = p.gx + (size_t)b * 3072;
    float hold = p.h_in_f[b * 1024 + u];
    float z  = sigmoidf_(gxrow[u]        + az[r] + b1z);
    float rr = sigmoidf_(gxrow[1024 + u] + ar[r] + b1r);
    float nn = tanhf(gxrow[2048 + u] + rr * (an[r] + b1n));
    float h2 = z * hold + (1.0f - z) * nn;
    p.h_out_f[b * 1024 + u]  = h2;
    p.h_out_bf[b * 1024 + u] = (bf16)h2;
    if (p.y) p.y[(size_t)b * 1024 + u] = (bf16)h2;
  }
}

// ---------------------------------------------------------------------------
// Fused Bahdanau attention for one decoder step. block per batch row b.
// score[t] = tanh(enc_proj[b,t,:] + q[b,:]) . V + bv; softmax over t;
// ctx = sum_t aw[t] * enc_out[b,t,:]; gin = [ctx | demb]  (bf16)
// ---------------------------------------------------------------------------
__global__ void __launch_bounds__(256)
attention_kernel(const float* __restrict__ q,
                 const float* __restrict__ enc_proj,   // [B][64][1024]
                 const bf16*  __restrict__ enc_out,    // [B][64][2048]
                 const float* __restrict__ attV,
                 const float* __restrict__ att_bv,
                 const bf16*  __restrict__ demb_s,     // [B][512]
                 bf16* __restrict__ gin) {             // [B][2560]
  __shared__ float qs[1024];
  __shared__ float part[64][4];
  __shared__ float sc[64];
  __shared__ float red;
  int b = blockIdx.x, tid = threadIdx.x;
  for (int i = tid; i < 1024; i += 256) qs[i] = q[b * 1024 + i];
  __syncthreads();
  {
    int t = tid & 63, qtr = tid >> 6;
    const float* ep = enc_proj + ((size_t)b * 64 + t) * 1024;
    float s = 0.0f;
    int k0 = qtr * 256;
    for (int k = k0; k < k0 + 256; ++k) s += tanhf(ep[k] + qs[k]) * attV[k];
    part[t][qtr] = s;
  }
  __syncthreads();
  if (tid < 64)
    sc[tid] = part[tid][0] + part[tid][1] + part[tid][2] + part[tid][3] + att_bv[0];
  __syncthreads();
  if (tid == 0) {
    float mx = sc[0];
    for (int t = 1; t < 64; ++t) mx = fmaxf(mx, sc[t]);
    red = mx;
  }
  __syncthreads();
  if (tid < 64) sc[tid] = __expf(sc[tid] - red);
  __syncthreads();
  if (tid == 0) {
    float s = 0.0f;
    for (int t = 0; t < 64; ++t) s += sc[t];
    red = 1.0f / s;
  }
  __syncthreads();
  if (tid < 64) sc[tid] *= red;
  __syncthreads();
  for (int c = tid; c < 2048; c += 256) {
    const bf16* eo = enc_out + (size_t)b * 64 * 2048 + c;
    float acc = 0.0f;
    for (int t = 0; t < 64; ++t) acc += sc[t] * (float)eo[t * 2048];
    gin[b * 2560 + c] = (bf16)acc;
  }
  for (int e = tid; e < 512; e += 256)
    gin[b * 2560 + 2048 + e] = demb_s[b * 512 + e];
}

// ---------------------------------------------------------------------------
// Small utility kernels
// ---------------------------------------------------------------------------
__global__ void transpose_bf16_kernel(const float* __restrict__ W,
                                      bf16* __restrict__ Wt, int K, int N,
                                      long total) {
  long i = (long)blockIdx.x * 256 + threadIdx.x;
  if (i >= total) return;
  int n = (int)(i / K), k = (int)(i % K);
  Wt[i] = (bf16)W[(size_t)k * N + n];
}

// out[t][b][e] = emb[tok[b][t]][e]   (B=64, E=512)
__global__ void embed_kernel(const int* __restrict__ tok,
                             const float* __restrict__ emb,
                             bf16* __restrict__ outp, int S, long total) {
  long i = (long)blockIdx.x * 256 + threadIdx.x;
  if (i >= total) return;
  int e = (int)(i % 512);
  long bt = i / 512;
  int b = (int)(bt % 64);
  int t = (int)(bt / 64);
  int row = tok[b * S + t];
  outp[i] = (bf16)emb[(size_t)row * 512 + e];
}

// enc_out[b][t][c] = c<1024 ? yf[t][b][c] : yb[t][b][c-1024]
__global__ void pack_encout_kernel(const bf16* __restrict__ yf,
                                   const bf16* __restrict__ yb,
                                   bf16* __restrict__ eo, long total) {
  long i = (long)blockIdx.x * 256 + threadIdx.x;
  if (i >= total) return;
  int c = (int)(i & 2047);
  long bt = i >> 11;
  int t = (int)(bt & 63);
  int b = (int)(bt >> 6);
  size_t src = ((size_t)t * 64 + b) * 1024;
  eo[i] = (c < 1024) ? yf[src + c] : yb[src + (c - 1024)];
}

__global__ void pack_hcat_kernel(const bf16* __restrict__ hf,
                                 const bf16* __restrict__ hb,
                                 bf16* __restrict__ hcat) {
  int i = blockIdx.x * 256 + threadIdx.x;  // 64*2048 total
  int c = i & 2047, b = i >> 11;
  hcat[i] = (c < 1024) ? hf[b * 1024 + c] : hb[b * 1024 + c - 1024];
}

__global__ void zero_kernel(float4* p, int n) {
  int i = blockIdx.x * 256 + threadIdx.x;
  if (i < n) p[i] = make_float4(0.f, 0.f, 0.f, 0.f);
}

// ---------------------------------------------------------------------------
// Host orchestration
// ---------------------------------------------------------------------------
extern "C" void kernel_launch(void* const* d_in, const int* in_sizes, int n_in,
                              void* d_out, int out_size, void* d_ws, size_t ws_size,
                              hipStream_t stream) {
  const int*   enc_tok = (const int*)d_in[0];
  const int*   dec_tok = (const int*)d_in[1];
  const float* enc_emb = (const float*)d_in[2];
  const float* ef_Wk = (const float*)d_in[3];
  const float* ef_Wr = (const float*)d_in[4];
  const float* ef_b  = (const float*)d_in[5];
  const float* eb_Wk = (const float*)d_in[6];
  const float* eb_Wr = (const float*)d_in[7];
  const float* eb_b  = (const float*)d_in[8];
  const float* efc_W = (const float*)d_in[9];
  const float* efc_b = (const float*)d_in[10];
  const float* dec_emb = (const float*)d_in[11];
  const float* att_W1 = (const float*)d_in[12];
  const float* att_b1 = (const float*)d_in[13];
  const float* att_W2 = (const float*)d_in[14];
  const float* att_b2 = (const float*)d_in[15];
  const float* att_V  = (const float*)d_in[16];
  const float* att_bv = (const float*)d_in[17];
  const float* dg_Wk = (const float*)d_in[18];
  const float* dg_Wr = (const float*)d_in[19];
  const float* dg_b  = (const float*)d_in[20];
  const float* fc1_W = (const float*)d_in[21];
  const float* fc1_b = (const float*)d_in[22];
  const float* fc2_W = (const float*)d_in[23];
  const float* fc2_b = (const float*)d_in[24];
  float* out = (float*)d_out;

  char* base = (char*)d_ws;
  size_t off = 0;
  auto alloc = [&](size_t bytes) -> void* {
    off = (off + 255) & ~(size_t)255;
    void* p = base + off;
    off += bytes;
    return p;
  };

  // bf16 transposed weights
  bf16* efWk_t = (bf16*)alloc(3072ull * 512 * 2);
  bf16* ebWk_t = (bf16*)alloc(3072ull * 512 * 2);
  bf16* efWr_t = (bf16*)alloc(3072ull * 1024 * 2);
  bf16* ebWr_t = (bf16*)alloc(3072ull * 1024 * 2);
  bf16* efc_t  = (bf16*)alloc(1024ull * 2048 * 2);
  bf16* aW1_t  = (bf16*)alloc(1024ull * 2048 * 2);
  bf16* aW2_t  = (bf16*)alloc(1024ull * 1024 * 2);
  bf16* dgWk_t = (bf16*)alloc(3072ull * 2560 * 2);
  bf16* dgWr_t = (bf16*)alloc(3072ull * 1024 * 2);
  bf16* fc1_t  = (bf16*)alloc(1024ull * 1024 * 2);
  bf16* fc2_t  = (bf16*)alloc(32000ull * 1024 * 2);
  // activations
  bf16*  xs_bf   = (bf16*)alloc(64ull * 64 * 512 * 2);   // [t][b][e]
  bf16*  demb_bf = (bf16*)alloc(32ull * 64 * 512 * 2);   // [s][b][e]
  float* gx_f    = (float*)alloc(64ull * 64 * 3072 * 4); // [t][b][3U]
  float* gx_b    = (float*)alloc(64ull * 64 * 3072 * 4);
  bf16*  y_f     = (bf16*)alloc(64ull * 64 * 1024 * 2);  // [t][b][u]
  bf16*  y_b     = (bf16*)alloc(64ull * 64 * 1024 * 2);
  bf16*  encout  = (bf16*)alloc(64ull * 64 * 2048 * 2);  // [b][t][c]
  float* encproj = (float*)alloc(64ull * 64 * 1024 * 4); // [b][t][u]
  float* h_enc_f = (float*)alloc(4ull * 64 * 1024 * 4);  // [dir][slot][b][u]
  bf16*  h_enc_b = (bf16*)alloc(4ull * 64 * 1024 * 2);
  bf16*  hcat    = (bf16*)alloc(64ull * 2048 * 2);
  float* h_dec_f = (float*)alloc(2ull * 64 * 1024 * 4);  // [slot][b][u]
  bf16*  h_dec_b = (bf16*)alloc(2ull * 64 * 1024 * 2);
  float* qbuf    = (float*)alloc(64ull * 1024 * 4);
  bf16*  gin     = (bf16*)alloc(64ull * 2560 * 2);
  float* gx_d    = (float*)alloc(64ull * 3072 * 4);
  bf16*  o_bf    = (bf16*)alloc(64ull * 1024 * 2);

  auto tconv = [&](const float* W, bf16* Wt, int K, int N) {
    long total = (long)K * N;
    transpose_bf16_kernel<<<dim3((unsigned)((total + 255) / 256)), dim3(256), 0, stream>>>(
        W, Wt, K, N, total);
  };
  auto gemm = [&](const bf16* A, int lda, const bf16* Bt, int ldb,
                  const float* bias, float* Cf, long ldcf, long cbase,
                  bf16* Cb, int ldcb, int M, int N, int K, int act) {
    gemm_kernel<<<dim3(N / 128, M / 64), dim3(32, 8), 0, stream>>>(
        A, lda, Bt, ldb, bias, Cf, ldcf, cbase, Cb, ldcb, K, act);
  };

  // 1) weight conversion (f32 -> bf16, transposed to [N][K])
  tconv(ef_Wk, efWk_t, 512, 3072);
  tconv(eb_Wk, ebWk_t, 512, 3072);
  tconv(ef_Wr, efWr_t, 1024, 3072);
  tconv(eb_Wr, ebWr_t, 1024, 3072);
  tconv(efc_W, efc_t, 2048, 1024);
  tconv(att_W1, aW1_t, 2048, 1024);
  tconv(att_W2, aW2_t, 1024, 1024);
  tconv(dg_Wk, dgWk_t, 2560, 3072);
  tconv(dg_Wr, dgWr_t, 1024, 3072);
  tconv(fc1_W, fc1_t, 1024, 1024);
  tconv(fc2_W, fc2_t, 1024, 32000);

  // 2) embedding gathers
  {
    long tot = 64L * 64 * 512;
    embed_kernel<<<dim3((unsigned)((tot + 255) / 256)), dim3(256), 0, stream>>>(
        enc_tok, enc_emb, xs_bf, 64, tot);
    long tot2 = 32L * 64 * 512;
    embed_kernel<<<dim3((unsigned)((tot2 + 255) / 256)), dim3(256), 0, stream>>>(
        dec_tok, dec_emb, demb_bf, 32, tot2);
  }

  // 3) zero encoder hidden state (both slots, both dirs; f32 + bf16)
  zero_kernel<<<dim3(4 * 64 * 1024 / 4 / 256), dim3(256), 0, stream>>>(
      (float4*)h_enc_f, 4 * 64 * 1024 / 4);
  zero_kernel<<<dim3(4 * 64 * 1024 * 2 / 16 / 256), dim3(256), 0, stream>>>(
      (float4*)h_enc_b, 4 * 64 * 1024 * 2 / 16);

  // 4) hoisted encoder input GEMMs: gx = xs @ Wk + b0   (M=4096,N=3072,K=512)
  gemm(xs_bf, 512, efWk_t, 512, ef_b, gx_f, 3072, 0, nullptr, 0, 4096, 3072, 512, 0);
  gemm(xs_bf, 512, ebWk_t, 512, eb_b, gx_b, 3072, 0, nullptr, 0, 4096, 3072, 512, 0);

  // 5) bidirectional GRU recurrence (64 stream-ordered steps, fwd+bwd fused)
  const size_t HS = 64 * 1024;
  for (int t = 0; t < 64; ++t) {
    int si = t & 1, so = (t + 1) & 1;
    GruDir pf = { h_enc_b + (0 * 2 + si) * HS, h_enc_f + (0 * 2 + si) * HS,
                  efWr_t, ef_b + 3072, gx_f + (size_t)t * 64 * 3072,
                  h_enc_f + (0 * 2 + so) * HS, h_enc_b + (0 * 2 + so) * HS,
                  y_f + (size_t)t * 64 * 1024 };
    GruDir pb = { h_enc_b + (1 * 2 + si) * HS, h_enc_f + (1 * 2 + si) * HS,
                  ebWr_t, eb_b + 3072, gx_b + (size_t)(63 - t) * 64 * 3072,
                  h_enc_f + (1 * 2 + so) * HS, h_enc_b + (1 * 2 + so) * HS,
                  y_b + (size_t)(63 - t) * 64 * 1024 };
    gru_step_kernel<<<dim3(64, 2), dim3(32, 4), 0, stream>>>(pf, pb);
  }

  // 6) hidden = concat(hf,hb) @ efc_W + b  (final h lives in slot 0)
  pack_hcat_kernel<<<dim3(64 * 2048 / 256), dim3(256), 0, stream>>>(
      h_enc_b + 0 * HS, h_enc_b + 2 * HS, hcat);
  gemm(hcat, 2048, efc_t, 2048, efc_b, h_dec_f, 1024, 0, h_dec_b, 1024,
       64, 1024, 2048, 0);

  // 7) enc_out pack + enc_proj = enc_out @ att_W1 + b1 (M=4096,N=1024,K=2048)
  {
    long tot = 64L * 64 * 2048;
    pack_encout_kernel<<<dim3((unsigned)((tot + 255) / 256)), dim3(256), 0, stream>>>(
        y_f, y_b, encout, tot);
  }
  gemm(encout, 2048, aW1_t, 2048, att_b1, encproj, 1024, 0, nullptr, 0,
       4096, 1024, 2048, 0);

  // 8) decoder: 32 stream-ordered steps
  for (int s = 0; s < 32; ++s) {
    int si = s & 1, so = (s + 1) & 1;
    // q = h @ att_W2 + b2
    gemm(h_dec_b + si * HS, 1024, aW2_t, 1024, att_b2, qbuf, 1024, 0,
         nullptr, 0, 64, 1024, 1024, 0);
    // attention -> gin = [ctx | demb]
    attention_kernel<<<dim3(64), dim3(256), 0, stream>>>(
        qbuf, encproj, encout, att_V, att_bv,
        demb_bf + (size_t)s * 64 * 512, gin);
    // gx_d = gin @ dg_Wk + b0  (M=64,N=3072,K=2560)
    gemm(gin, 2560, dgWk_t, 2560, dg_b, gx_d, 3072, 0, nullptr, 0,
         64, 3072, 2560, 0);
    // decoder GRU step
    GruDir pd = { h_dec_b + si * HS, h_dec_f + si * HS, dgWr_t, dg_b + 3072,
                  gx_d, h_dec_f + so * HS, h_dec_b + so * HS, nullptr };
    gru_step_kernel<<<dim3(64, 1), dim3(32, 4), 0, stream>>>(pd, pd);
    // o = relu(h2 @ fc1 + b)
    gemm(h_dec_b + so * HS, 1024, fc1_t, 1024, fc1_b, nullptr, 0, 0,
         o_bf, 1024, 64, 1024, 1024, 1);
    // logits[b][s][:] = o @ fc2 + b   -> strided into d_out
    gemm(o_bf, 1024, fc2_t, 1024, fc2_b, out, 32L * 32000, (long)s * 32000,
         nullptr, 0, 64, 32000, 1024, 0);
  }
  (void)in_sizes; (void)n_in; (void)out_size; (void)ws_size;
}
